// LinearCrossAttention_13726715478155
// MI455X (gfx1250) — matrix operand order repaired and checked by reading, not compiled, not run on hardware
//
#include <hip/hip_runtime.h>
#include <hip/hip_bf16.h>
#include <math.h>

typedef __attribute__((ext_vector_type(16))) _Float16 v16h;
typedef __attribute__((ext_vector_type(8)))  float    v8f;
typedef int v4i_t __attribute__((vector_size(16)));
typedef __attribute__((address_space(1))) v4i_t* gv4i_p;
typedef __attribute__((address_space(3))) v4i_t* lv4i_p;

#define N_EMBD 256
#define N_HEAD 8
#define HDIM   32
#define BB     4
#define TT     8192
#define MROWS  (BB * TT)          // 32768 rows for x / per-input y
#define WFRAG  512                // halves per packed B fragment (32 lanes * 16)
#define WMAT   (128 * WFRAG)      // halves per packed 256x256 matrix

#if __has_builtin(__builtin_amdgcn_global_load_async_to_lds_b128)
#define HAVE_ASYNC_LDS 1
#else
#define HAVE_ASYNC_LDS 0
#endif

// 16-byte global -> LDS copy, async (ASYNCcnt / CDNA5 path) when available
__device__ __forceinline__ void copy16_g2l(const _Float16* __restrict__ src,
                                           _Float16* __restrict__ dstLds) {
#if HAVE_ASYNC_LDS
  __builtin_amdgcn_global_load_async_to_lds_b128(
      (gv4i_p)const_cast<_Float16*>(src), (lv4i_p)dstLds, 0, 0);
#else
  *((float4*)dstLds) = *((const float4*)src);
#endif
}

__device__ __forceinline__ void async_join() {
#if HAVE_ASYNC_LDS
#if __has_builtin(__builtin_amdgcn_s_wait_asynccnt)
  __builtin_amdgcn_s_wait_asynccnt(0);
#else
  asm volatile("s_wait_asynccnt 0x0" ::: "memory");
#endif
#endif
  __syncthreads();
}

// ---------------------------------------------------------------------------
// WMMA fragment helpers (wave32, 16x16x32 f16 -> f32)
// A 16x32 (f16): lane<16 -> M=lane,    halves j : K=j
//                lane>=16-> M=lane-16, halves j : K=16+j
// ---------------------------------------------------------------------------
__device__ __forceinline__ v16h load_frag_a_f32(const float* __restrict__ base,
                                                int ld, int lane) {
  const int m  = lane & 15;
  const int kb = (lane >> 4) << 4;          // 0 or 16
  const float* p = base + (size_t)m * ld + kb;
  const float4 x0 = *(const float4*)(p);
  const float4 x1 = *(const float4*)(p + 4);
  const float4 x2 = *(const float4*)(p + 8);
  const float4 x3 = *(const float4*)(p + 12);
  v16h a;
  a[0]  = (_Float16)x0.x; a[1]  = (_Float16)x0.y;
  a[2]  = (_Float16)x0.z; a[3]  = (_Float16)x0.w;
  a[4]  = (_Float16)x1.x; a[5]  = (_Float16)x1.y;
  a[6]  = (_Float16)x1.z; a[7]  = (_Float16)x1.w;
  a[8]  = (_Float16)x2.x; a[9]  = (_Float16)x2.y;
  a[10] = (_Float16)x2.z; a[11] = (_Float16)x2.w;
  a[12] = (_Float16)x3.x; a[13] = (_Float16)x3.y;
  a[14] = (_Float16)x3.z; a[15] = (_Float16)x3.w;
  return a;
}

// Packed B fragment resident in LDS: 16 contiguous halves/lane -> 2x ds_load_b128
__device__ __forceinline__ v16h load_frag_b_ldspk(const _Float16* __restrict__ P,
                                                  int f, int lane) {
  return *((const v16h*)(P + (size_t)f * WFRAG + lane * 16));
}

__device__ __forceinline__ v16h load_frag_b_lds(const _Float16* __restrict__ base,
                                                int ld, int lane) {
  const int n  = lane & 15;
  const int kb = (lane >> 4) << 4;
  const _Float16* p = base + kb * ld + n;
  v16h b;
#pragma unroll
  for (int j = 0; j < 16; ++j) b[j] = p[j * ld];
  return b;
}

// A = transpose of a [32 x 16] LDS block: A[mi,kj] = lds[kj*ld + col0 + mi]
__device__ __forceinline__ v16h load_frag_aT_lds(const _Float16* __restrict__ lds,
                                                 int ld, int col0, int lane) {
  const int mi = lane & 15;
  const int kb = (lane >> 4) << 4;
  v16h a;
#pragma unroll
  for (int j = 0; j < 8; ++j) {
    a[j]     = lds[(kb + j) * ld + col0 + mi];
    a[j + 8] = lds[(8 + kb + j) * ld + col0 + mi];
  }
  return a;
}

// C 16x16 f32: lane<16 -> N=lane, rows 0..7 ; lane>=16 -> N=lane-16, rows 8..15
__device__ __forceinline__ void store_frag_c(float* __restrict__ base, int ld,
                                             int lane, v8f c,
                                             const float* __restrict__ bias) {
  const int n  = lane & 15;
  const int mb = (lane >> 4) << 3;
  const float bv = bias[n];
#pragma unroll
  for (int r = 0; r < 8; ++r) base[(size_t)(mb + r) * ld + n] = c[r] + bv;
}

__device__ __forceinline__ void store_frag_c_lds(_Float16* __restrict__ base, int ld,
                                                 int lane, v8f c,
                                                 const float* __restrict__ bias) {
  const int n  = lane & 15;
  const int mb = (lane >> 4) << 3;
  const float bv = bias[n];
#pragma unroll
  for (int r = 0; r < 8; ++r) base[(mb + r) * ld + n] = (_Float16)(c[r] + bv);
}

// ---------------------------------------------------------------------------
// Pack one 256x256 f32 weight matrix into fragment-contiguous f16:
//   P[f*512 + lane*16 + j], f = ks*16 + nt (B-fragment layout).
// ---------------------------------------------------------------------------
__global__ void pack_w_kernel(const float* __restrict__ W, _Float16* __restrict__ P) {
  const int t  = blockIdx.x * blockDim.x + threadIdx.x;   // 0..4095
  const int f  = t >> 5;                                  // 0..127
  const int l  = t & 31;
  const int ks = f >> 4;
  const int nt = f & 15;
  const int n  = l & 15;
  const int kb = (l >> 4) << 4;
  const float* src = W + (size_t)(ks * 32 + kb) * N_EMBD + nt * 16 + n;
  v16h v;
#pragma unroll
  for (int j = 0; j < 16; ++j) v[j] = (_Float16)src[(size_t)j * N_EMBD];
  *((v16h*)(P + (size_t)f * WFRAG + l * 16)) = v;
}

// ---------------------------------------------------------------------------
// GEMM: C[M,256] = A[M,256] * W[256,256] + bias   (f16 WMMA, f32 accum)
// grid = (M/128, 2); block = 256 (8 waves).  Block covers 128 rows x 128 cols
// (column half = blockIdx.y); its 64KB half of packed W staged in LDS (async).
// ---------------------------------------------------------------------------
__global__ void gemm256_kernel(const float* __restrict__ A,
                               const _Float16* __restrict__ Wp16,
                               const float* __restrict__ bias,
                               float* __restrict__ C) {
  extern __shared__ _Float16 ldsW[];                // 64 frags * 512 = 64KB
  const int lane = threadIdx.x & 31;
  const int wave = threadIdx.x >> 5;
  const int ch   = blockIdx.y;                      // column half
  const int row0 = blockIdx.x * 128 + wave * 16;
  const float* arow = A + (size_t)row0 * N_EMBD;

  // stage this block's weight half: 4096 x 16B chunks, async to LDS
  for (int c = threadIdx.x; c < 4096; c += 256) {
    const int ks  = c >> 9;
    const int off = (c & 511) * 8;                  // halves within 8-frag span
    copy16_g2l(Wp16 + ((size_t)(ks * 16 + ch * 8) * WFRAG) + off,
               ldsW + (size_t)c * 8);
  }
  async_join();

  v8f acc[8] = {};
#pragma unroll
  for (int ks = 0; ks < 8; ++ks) {
    v16h a = load_frag_a_f32(arow + ks * 32, N_EMBD, lane);
    if (ks < 7) {  // prefetch next K-step of the streaming operand
      __builtin_prefetch(arow + (size_t)(lane & 15) * N_EMBD + ks * 32 + 32, 0, 3);
    }
#pragma unroll
    for (int nt = 0; nt < 8; ++nt) {
      v16h bfr = load_frag_b_ldspk(ldsW, ks * 8 + nt, lane);
      acc[nt] = __builtin_amdgcn_wmma_f32_16x16x32_f16(
          false, a, false, bfr, (short)0, acc[nt], false, false);
    }
  }
#pragma unroll
  for (int nt = 0; nt < 8; ++nt) {
    const int n0 = ch * 128 + nt * 16;
    store_frag_c(C + (size_t)row0 * N_EMBD + n0, N_EMBD, lane, acc[nt], bias + n0);
  }
}

// ---------------------------------------------------------------------------
// In-place softmax over each 32-column head block; one thread per (row, head)
// ---------------------------------------------------------------------------
__global__ void softmax32_kernel(float* __restrict__ P, int nRowHead) {
  const int idx = blockIdx.x * blockDim.x + threadIdx.x;
  if (idx >= nRowHead) return;
  float* p = P + (size_t)(idx >> 3) * N_EMBD + (idx & 7) * HDIM;
  float vals[HDIM];
  float m = -3.4e38f;
#pragma unroll
  for (int j = 0; j < HDIM; ++j) { vals[j] = p[j]; m = fmaxf(m, vals[j]); }
  float s = 0.f;
#pragma unroll
  for (int j = 0; j < HDIM; ++j) { vals[j] = __expf(vals[j] - m); s += vals[j]; }
  const float r = 1.f / s;
#pragma unroll
  for (int j = 0; j < HDIM; ++j) p[j] = vals[j] * r;
}

// ---------------------------------------------------------------------------
// Fused K/V kernel for one input i, split per column half (4 heads/block):
//  - stage this half of packed Wk and Wv in LDS (async, 128KB)
//  - project 64 rows of y -> K,V (64x128 f16 tiles in LDS, WMMA)
//  - softmax K per (row, head); accumulate ksum (atomic)
//  - per head: kv[b,h] (32x32) += K_h^T * V_h over 64 rows (WMMA + atomics)
// grid = (M/64, 2); block = 256 (8 waves); dynamic LDS = 160KB (2 blocks/WGP).
// ---------------------------------------------------------------------------
__global__ void kv_kernel(const float* __restrict__ Y,
                          const _Float16* __restrict__ pWk,
                          const float* __restrict__ bk,
                          const _Float16* __restrict__ pWv,
                          const float* __restrict__ bv,
                          float* __restrict__ ksum,
                          float* __restrict__ kvb) {
  extern __shared__ _Float16 smem[];
  _Float16* ldsK  = smem;                           // 64*128
  _Float16* ldsV  = smem + 64 * 128;                // 64*128
  _Float16* ldsWk = smem + 2 * 64 * 128;            // 64 frags
  _Float16* ldsWv = ldsWk + 64 * WFRAG;             // 64 frags

  const int lane = threadIdx.x & 31;
  const int wave = threadIdx.x >> 5;
  const int ch   = blockIdx.y;                      // column half (heads 4ch..4ch+3)
  const int row0 = blockIdx.x * 64;
  const int b    = row0 >> 13;                      // row0 / 8192

  // stage Wk-half and Wv-half: 8192 x 16B chunks, async to LDS
  for (int c = threadIdx.x; c < 8192; c += 256) {
    const int cc  = c & 4095;
    const int ks  = cc >> 9;
    const int off = (cc & 511) * 8;
    const _Float16* src =
        ((c < 4096) ? pWk : pWv) + ((size_t)(ks * 16 + ch * 8) * WFRAG) + off;
    _Float16* dst = ((c < 4096) ? ldsWk : ldsWv) + (size_t)cc * 8;
    copy16_g2l(src, dst);
  }
  async_join();

  // ---- phase 1: K and V projections of 64 rows x 128 cols into LDS ----
  {
    const int isV = wave & 1;
    const int mt  = wave >> 1;                      // 4 row tiles of 16
    const _Float16* Wl = isV ? ldsWv : ldsWk;
    const float*    bm = (isV ? bv : bk) + ch * 128;
    _Float16*      out = isV ? ldsV : ldsK;
    const float*  arow = Y + (size_t)(row0 + mt * 16) * N_EMBD;

    v8f acc[8] = {};
#pragma unroll
    for (int ks = 0; ks < 8; ++ks) {
      v16h a = load_frag_a_f32(arow + ks * 32, N_EMBD, lane);
      if (ks < 7) {
        __builtin_prefetch(arow + (size_t)(lane & 15) * N_EMBD + ks * 32 + 32, 0, 3);
      }
#pragma unroll
      for (int nt = 0; nt < 8; ++nt) {
        v16h bfr = load_frag_b_ldspk(Wl, ks * 8 + nt, lane);
        acc[nt] = __builtin_amdgcn_wmma_f32_16x16x32_f16(
            false, a, false, bfr, (short)0, acc[nt], false, false);
      }
    }
#pragma unroll
    for (int nt = 0; nt < 8; ++nt) {
      store_frag_c_lds(out + (mt * 16) * 128 + nt * 16, 128, lane, acc[nt],
                       bm + nt * 16);
    }
  }
  __syncthreads();

  // ---- phase 2: softmax K per (row, head): 64 rows * 4 heads = 256 threads
  {
    const int row = threadIdx.x >> 2;
    const int hh  = threadIdx.x & 3;
    _Float16* p = ldsK + row * 128 + hh * HDIM;
    float vals[HDIM];
    float m = -3.4e38f;
#pragma unroll
    for (int j = 0; j < HDIM; ++j) { vals[j] = (float)p[j]; m = fmaxf(m, vals[j]); }
    float s = 0.f;
#pragma unroll
    for (int j = 0; j < HDIM; ++j) { vals[j] = __expf(vals[j] - m); s += vals[j]; }
    const float r = 1.f / s;
#pragma unroll
    for (int j = 0; j < HDIM; ++j) p[j] = (_Float16)(vals[j] * r);
  }
  __syncthreads();

  // ---- phase 3: ksum column partial sums (2 threads per column) ----
  {
    const int c  = threadIdx.x & 127;
    const int r0 = (threadIdx.x >> 7) * 32;
    float s = 0.f;
#pragma unroll
    for (int r = 0; r < 32; ++r) s += (float)ldsK[(r0 + r) * 128 + c];
    atomicAdd(&ksum[b * N_EMBD + ch * 128 + c], s);
  }

  // ---- phase 4: kv[b,h] += K_h^T * V_h over 64 rows (wave = (head, mt2)) ----
  {
    const int hh  = wave >> 1;                      // head within half
    const int mt2 = wave & 1;                       // d-row tile of kv
    const int h   = ch * 4 + hh;
    float* kvh = kvb + ((size_t)(b * N_HEAD + h)) * (HDIM * HDIM);
    v8f acc2[2] = {};
#pragma unroll
    for (int ks2 = 0; ks2 < 2; ++ks2) {
      v16h a = load_frag_aT_lds(ldsK + ks2 * 32 * 128, 128,
                                hh * HDIM + mt2 * 16, lane);
#pragma unroll
      for (int nt2 = 0; nt2 < 2; ++nt2) {
        v16h bfr = load_frag_b_lds(ldsV + ks2 * 32 * 128 + hh * HDIM + nt2 * 16,
                                   128, lane);
        acc2[nt2] = __builtin_amdgcn_wmma_f32_16x16x32_f16(
            false, a, false, bfr, (short)0, acc2[nt2], false, false);
      }
    }
    const int n  = lane & 15;
    const int mb = (lane >> 4) << 3;
#pragma unroll
    for (int nt2 = 0; nt2 < 2; ++nt2)
#pragma unroll
      for (int r = 0; r < 8; ++r)
        atomicAdd(&kvh[(mt2 * 16 + mb + r) * HDIM + nt2 * 16 + n], acc2[nt2][r]);
  }
}

// ---------------------------------------------------------------------------
// Combine: row-wise, in place over Q:
//   out[c] = q[c] + Dinv0 * (q_h . kv0_h[:,e]) + Dinv1 * (q_h . kv1_h[:,e])
// ---------------------------------------------------------------------------
__global__ void combine_kernel(float* __restrict__ Q,
                               const float* __restrict__ ksum0,
                               const float* __restrict__ ksum1,
                               const float* __restrict__ kv0,
                               const float* __restrict__ kv1) {
  const int lane = threadIdx.x & 31;
  const int warp = threadIdx.x >> 5;
  const size_t row = (size_t)blockIdx.x * 8 + warp;
  const int b = (int)(row >> 13);
  float* qrow = Q + row * N_EMBD;

#pragma unroll
  for (int h = 0; h < N_HEAD; ++h) {
    const float qv = qrow[h * HDIM + lane];
    const float k0 = ksum0[b * N_EMBD + h * HDIM + lane];
    const float k1 = ksum1[b * N_EMBD + h * HDIM + lane];
    float t0 = qv * k0;
    float t1 = qv * k1;
#pragma unroll
    for (int off = 16; off > 0; off >>= 1) {
      t0 += __shfl_xor(t0, off, 32);
      t1 += __shfl_xor(t1, off, 32);
    }
    const float Dinv0 = 1.f / t0;
    const float Dinv1 = 1.f / t1;
    const float* kvh0 = kv0 + ((size_t)(b * N_HEAD + h)) * (HDIM * HDIM);
    const float* kvh1 = kv1 + ((size_t)(b * N_HEAD + h)) * (HDIM * HDIM);
    float s0 = 0.f, s1 = 0.f;
#pragma unroll
    for (int d = 0; d < HDIM; ++d) {
      const float qd = __shfl(qv, d, 32);
      s0 += qd * kvh0[d * HDIM + lane];
      s1 += qd * kvh1[d * HDIM + lane];
    }
    qrow[h * HDIM + lane] = qv + Dinv0 * s0 + Dinv1 * s1;
  }
}

__global__ void zero_kernel(float* __restrict__ p, int n) {
  const int i = blockIdx.x * blockDim.x + threadIdx.x;
  if (i < n) p[i] = 0.f;
}

// ---------------------------------------------------------------------------
extern "C" void kernel_launch(void* const* d_in, const int* in_sizes, int n_in,
                              void* d_out, int out_size, void* d_ws, size_t ws_size,
                              hipStream_t stream) {
  const float* x  = (const float*)d_in[0];
  const float* y  = (const float*)d_in[1];
  const float* Wq = (const float*)d_in[2];
  const float* bq = (const float*)d_in[3];
  const float* Wk = (const float*)d_in[4];
  const float* bk = (const float*)d_in[5];
  const float* Wv = (const float*)d_in[6];
  const float* bv = (const float*)d_in[7];
  const float* Wp = (const float*)d_in[8];
  const float* bp = (const float*)d_in[9];
  float* out = (float*)d_out;

  // workspace layout
  float* Q     = (float*)d_ws;                       // MROWS*256 f32 (in-place reuse)
  float* ksum0 = Q + (size_t)MROWS * N_EMBD;
  float* ksum1 = ksum0 + BB * N_EMBD;
  float* kv0   = ksum1 + BB * N_EMBD;
  float* kv1   = kv0 + BB * N_HEAD * HDIM * HDIM;
  _Float16* pW = (_Float16*)(kv1 + BB * N_HEAD * HDIM * HDIM);  // 6 packed mats
  _Float16* pWq  = pW;
  _Float16* pWk0 = pW + 1 * (size_t)WMAT;
  _Float16* pWk1 = pW + 2 * (size_t)WMAT;
  _Float16* pWv0 = pW + 3 * (size_t)WMAT;
  _Float16* pWv1 = pW + 4 * (size_t)WMAT;
  _Float16* pWp  = pW + 5 * (size_t)WMAT;

  // pack weights to fragment-contiguous f16 (tiny, L2-resident afterwards)
  pack_w_kernel<<<dim3(16), dim3(256), 0, stream>>>(Wq, pWq);
  pack_w_kernel<<<dim3(16), dim3(256), 0, stream>>>(Wk, pWk0);
  pack_w_kernel<<<dim3(16), dim3(256), 0, stream>>>(Wk + N_EMBD * N_EMBD, pWk1);
  pack_w_kernel<<<dim3(16), dim3(256), 0, stream>>>(Wv, pWv0);
  pack_w_kernel<<<dim3(16), dim3(256), 0, stream>>>(Wv + N_EMBD * N_EMBD, pWv1);
  pack_w_kernel<<<dim3(16), dim3(256), 0, stream>>>(Wp, pWp);

  const int nAccum = 2 * BB * N_EMBD + 2 * BB * N_HEAD * HDIM * HDIM;
  zero_kernel<<<dim3((nAccum + 255) / 256), dim3(256), 0, stream>>>(ksum0, nAccum);

  // Q projection + softmax
  gemm256_kernel<<<dim3(MROWS / 128, 2), dim3(256), 64 * 1024, stream>>>(
      x, pWq, bq, Q);
  softmax32_kernel<<<dim3(MROWS * N_HEAD / 256), dim3(256), 0, stream>>>(
      Q, MROWS * N_HEAD);

  // fused K/V projection + softmax + ksum + K^T V, per input
  const size_t ysz = (size_t)MROWS * N_EMBD;
  kv_kernel<<<dim3(MROWS / 64, 2), dim3(256), 160 * 1024, stream>>>(
      y, pWk0, bk, pWv0, bv, ksum0, kv0);
  kv_kernel<<<dim3(MROWS / 64, 2), dim3(256), 160 * 1024, stream>>>(
      y + ysz, pWk1, bk + N_EMBD, pWv1, bv + N_EMBD, ksum1, kv1);

  // combine (in place over Q)
  combine_kernel<<<dim3(MROWS / 8), dim3(256), 0, stream>>>(
      Q, ksum0, ksum1, kv0, kv1);

  // final projection -> d_out
  gemm256_kernel<<<dim3(MROWS / 128, 2), dim3(256), 64 * 1024, stream>>>(
      Q, pWp, bp, out);
}
